// MoELayer_23897198035630
// MI455X (gfx1250) — matrix operand order repaired and checked by reading, not compiled, not run on hardware
//
#include <hip/hip_runtime.h>
#include <math.h>

// ---------------------------------------------------------------------------
// Dense MoE on MI455X (gfx1250, wave32).
//   Pre-pass (memory-bound):  x -> bf16;  W -> bf16 transposed [N,K]
//   xh  = relu(x @ Wr1 + br1)        bf16 out          (WMMA)
//   rw  = softmax(xh @ Wr2 + br2)                      (wave/token reduction)
//   h_e = gelu(x @ W1[e] + b1[e])    bf16 out          (WMMA)
//   out (+)= rw[:,e]*(h_e @ W2[e] + b2[e])  fp32 out   (WMMA)
// GEMM: 128x128x32 tiles, 8 waves (4Mx2N), 8 v_wmma_f32_16x16x32_bf16 per
// wave per k-step, double-buffered LDS fed by global_load_async_to_lds_b128.
// ---------------------------------------------------------------------------

typedef __attribute__((ext_vector_type(16))) __bf16 v16bf;
typedef __attribute__((ext_vector_type(8)))  float  v8f;

#define BM 128
#define BN 128
#define BK 32
#define LDS_K 40   // LDS row pitch in bf16 elems (80 B): 16B-aligned chunks, bank-spread

union Frag { v16bf v; uint4 q[2]; };

__device__ __forceinline__ unsigned short f2bf(float f) {
    unsigned u = __float_as_uint(f);
    u += 0x7FFFu + ((u >> 16) & 1u);      // round-to-nearest-even
    return (unsigned short)(u >> 16);
}
__device__ __forceinline__ float bf2f(unsigned short h) {
    return __uint_as_float(((unsigned)h) << 16);
}

#if __has_builtin(__builtin_amdgcn_global_load_async_to_lds_b128)
#define HAVE_ASYNC_LDS 1
// exact builtin parameter types per hipcc diagnostic: vector_size(16) int,
// AS1 (global) source, AS3 (LDS) destination, imm offset, imm cpol
typedef int v4i_vs __attribute__((vector_size(16)));
typedef __attribute__((address_space(1))) v4i_vs* gbl_v4i_p;
typedef __attribute__((address_space(3))) v4i_vs* lds_v4i_p;
__device__ __forceinline__ void asyncCopy16(const unsigned short* g, unsigned short* l) {
    __builtin_amdgcn_global_load_async_to_lds_b128((gbl_v4i_p)g, (lds_v4i_p)l, 0, 0);
}
#else
#define HAVE_ASYNC_LDS 0
#endif

#if __has_builtin(__builtin_amdgcn_s_wait_asynccnt)
#define WAIT_ASYNC(n) __builtin_amdgcn_s_wait_asynccnt(n)
#else
#define WAIT_ASYNC(n) asm volatile("s_wait_asynccnt %0" ::"n"(n) : "memory")
#endif

// epi: 0 = bias+ReLU -> bf16 C, 1 = bias+GELU -> bf16 C, 2 = bias, *rw[:,ecol] -> fp32 C (+=)
__global__ __launch_bounds__(256) void gemm_bf16_wmma(
    const unsigned short* __restrict__ A,   // [M,K] bf16 row-major
    const unsigned short* __restrict__ Bt,  // [N,K] bf16 (pre-transposed weight)
    const float* __restrict__ bias,         // [N]
    void* __restrict__ Cv,                  // bf16 [M,N] (epi 0/1) or fp32 [M,N] (epi 2)
    const float* __restrict__ rw,           // [M,8] router weights (epi 2)
    int ecol, int accum, int N, int K, int epi)
{
    __shared__ __align__(16) unsigned short As[2][BM * LDS_K];  // [m][k] 2x10KB
    __shared__ __align__(16) unsigned short Bs[2][BN * LDS_K];  // [n][k] 2x10KB

    const int tid   = threadIdx.x;
    const int mBase = blockIdx.y * BM;
    const int nBase = blockIdx.x * BN;
    const int wave  = tid >> 5, lane = tid & 31;
    const int wm    = wave & 3, wn = wave >> 2;  // 4 waves on M, 2 on N
    const int lr    = lane & 15, half = lane >> 4;

    v8f acc[2][4];
#pragma unroll
    for (int i = 0; i < 2; ++i)
#pragma unroll
        for (int j = 0; j < 4; ++j)
#pragma unroll
            for (int v = 0; v < 8; ++v) acc[i][j][v] = 0.0f;

    // ---- one k-step of WMMA work from LDS buffer `buf` ----
    auto computeStep = [&](int buf) {
        Frag af[2], bfr[4];
#pragma unroll
        for (int i = 0; i < 2; ++i) {               // A frags: 16-bit A(16x32) layout
            int row = wm * 32 + i * 16 + lr;
            const unsigned short* p = &As[buf][row * LDS_K + half * 8];
            af[i].q[0] = *reinterpret_cast<const uint4*>(p);        // K = half*8 .. +7
            af[i].q[1] = *reinterpret_cast<const uint4*>(p + 16);   // K = 16+half*8 ..
        }
#pragma unroll
        for (int j = 0; j < 4; ++j) {               // B frags: 16-bit B(32x16) layout
            int col = wn * 64 + j * 16 + lr;
            const unsigned short* p = &Bs[buf][col * LDS_K + half * 8];
            bfr[j].q[0] = *reinterpret_cast<const uint4*>(p);       // K = half*8 .. +7
            bfr[j].q[1] = *reinterpret_cast<const uint4*>(p + 16);  // K = 16+half*8 ..
        }
#pragma unroll
        for (int i = 0; i < 2; ++i)
#pragma unroll
            for (int j = 0; j < 4; ++j)
                acc[i][j] = __builtin_amdgcn_wmma_f32_16x16x32_bf16(
                    false, af[i].v, false, bfr[j].v, (short)0, acc[i][j], false, false);
    };

#if HAVE_ASYNC_LDS
    // ---- async global->LDS staging, double buffered (4 asyncs/thread/tile) ----
    auto issueTile = [&](int k0, int buf) {
#pragma unroll
        for (int j = 0; j < 2; ++j) {
            int c = j * 256 + tid;                  // 512 chunks of 16 B per tile
            int row = c >> 2, q = c & 3;
            asyncCopy16(A  + (size_t)(mBase + row) * K + k0 + q * 8,
                        &As[buf][row * LDS_K + q * 8]);
            asyncCopy16(Bt + (size_t)(nBase + row) * K + k0 + q * 8,
                        &Bs[buf][row * LDS_K + q * 8]);
        }
    };
    issueTile(0, 0);
    int buf = 0;
    for (int k0 = 0; k0 < K; k0 += BK, buf ^= 1) {
        bool hasNext = (k0 + BK) < K;
        if (hasNext) { issueTile(k0 + BK, buf ^ 1); WAIT_ASYNC(4); }
        else         { WAIT_ASYNC(0); }
        __syncthreads();          // this tile visible to all waves
        computeStep(buf);
        __syncthreads();          // all readers done before buf is refilled
    }
#else
    // ---- fallback: synchronous b128 staging through VGPRs ----
    uint4 ar[2], br_[2];
    auto loadRegs = [&](int k0) {
#pragma unroll
        for (int j = 0; j < 2; ++j) {
            int c = j * 256 + tid;
            int row = c >> 2, q = c & 3;
            ar[j]  = *reinterpret_cast<const uint4*>(A  + (size_t)(mBase + row) * K + k0 + q * 8);
            br_[j] = *reinterpret_cast<const uint4*>(Bt + (size_t)(nBase + row) * K + k0 + q * 8);
        }
    };
    auto storeRegs = [&](int buf) {
#pragma unroll
        for (int j = 0; j < 2; ++j) {
            int c = j * 256 + tid;
            int row = c >> 2, q = c & 3;
            *reinterpret_cast<uint4*>(&As[buf][row * LDS_K + q * 8]) = ar[j];
            *reinterpret_cast<uint4*>(&Bs[buf][row * LDS_K + q * 8]) = br_[j];
        }
    };
    loadRegs(0);
    int buf = 0;
    for (int k0 = 0; k0 < K; k0 += BK, buf ^= 1) {
        storeRegs(buf);
        if (k0 + BK < K) loadRegs(k0 + BK);
        __syncthreads();
        computeStep(buf);
        __syncthreads();
    }
#endif

    // ---- epilogue ----
#pragma unroll
    for (int i = 0; i < 2; ++i) {
        int rowBase = mBase + wm * 32 + i * 16 + half * 8;
#pragma unroll
        for (int j = 0; j < 4; ++j) {
            int col = nBase + wn * 64 + j * 16 + lr;
            float bv = bias[col];
#pragma unroll
            for (int v = 0; v < 8; ++v) {
                int row = rowBase + v;
                float c = acc[i][j][v] + bv;
                if (epi == 0) {
                    ((unsigned short*)Cv)[(size_t)row * N + col] = f2bf(fmaxf(c, 0.0f));
                } else if (epi == 1) {
                    c = 0.5f * c * (1.0f + erff(c * 0.70710678118654752f));  // exact GELU
                    ((unsigned short*)Cv)[(size_t)row * N + col] = f2bf(c);
                } else {
                    float* C = (float*)Cv;
                    c *= rw[row * 8 + ecol];
                    if (accum) c += C[(size_t)row * N + col];
                    C[(size_t)row * N + col] = c;
                }
            }
        }
    }
}

// fp32 -> bf16 elementwise (x), float4-wide
__global__ __launch_bounds__(256) void cvt_bf16(
    const float* __restrict__ in, unsigned short* __restrict__ out, int n4)
{
    int i = blockIdx.x * 256 + threadIdx.x;
    if (i >= n4) return;
    float4 v = reinterpret_cast<const float4*>(in)[i];
    union { unsigned short s[4]; uint2 u; } r;
    r.s[0] = f2bf(v.x); r.s[1] = f2bf(v.y); r.s[2] = f2bf(v.z); r.s[3] = f2bf(v.w);
    reinterpret_cast<uint2*>(out)[i] = r.u;
}

// fp32 [K,N] -> bf16 [N,K], batched over blockIdx.z; 32x32 LDS tiles
__global__ __launch_bounds__(256) void transpose_cvt(
    const float* __restrict__ in, unsigned short* __restrict__ out, int K, int N)
{
    __shared__ float tile[32][33];
    size_t zo = (size_t)blockIdx.z * K * N;
    int k0 = blockIdx.y * 32, n0 = blockIdx.x * 32;
    int tx = threadIdx.x, ty = threadIdx.y;
#pragma unroll
    for (int j = 0; j < 32; j += 8)
        tile[ty + j][tx] = in[zo + (size_t)(k0 + ty + j) * N + n0 + tx];
    __syncthreads();
#pragma unroll
    for (int j = 0; j < 32; j += 8)
        out[zo + (size_t)(n0 + ty + j) * K + k0 + tx] = f2bf(tile[tx][ty + j]);
}

// Router second layer + softmax: one wave32 per token
__global__ __launch_bounds__(256) void router_softmax(
    const unsigned short* __restrict__ xh,  // [T,D] bf16 relu hidden
    const float* __restrict__ Wr2,          // [D,8]
    const float* __restrict__ br2,          // [8]
    float* __restrict__ rw,                 // [T,8]
    int T, int D)
{
    int wave = threadIdx.x >> 5, lane = threadIdx.x & 31;
    int t = blockIdx.x * 8 + wave;
    if (t >= T) return;

    float a[8];
#pragma unroll
    for (int e = 0; e < 8; ++e) a[e] = 0.0f;
    for (int d = lane; d < D; d += 32) {
        float xv = bf2f(xh[(size_t)t * D + d]);
        const float* w = Wr2 + (size_t)d * 8;
#pragma unroll
        for (int e = 0; e < 8; ++e) a[e] = fmaf(xv, w[e], a[e]);
    }
#pragma unroll
    for (int e = 0; e < 8; ++e)
#pragma unroll
        for (int off = 16; off; off >>= 1)
            a[e] += __shfl_xor(a[e], off, 32);

    if (lane == 0) {
        float m = -INFINITY;
#pragma unroll
        for (int e = 0; e < 8; ++e) { a[e] += br2[e]; m = fmaxf(m, a[e]); }
        float s = 0.0f;
#pragma unroll
        for (int e = 0; e < 8; ++e) { a[e] = __expf(a[e] - m); s += a[e]; }
        float inv = 1.0f / s;
#pragma unroll
        for (int e = 0; e < 8; ++e) rw[(size_t)t * 8 + e] = a[e] * inv;
    }
}

extern "C" void kernel_launch(void* const* d_in, const int* in_sizes, int n_in,
                              void* d_out, int out_size, void* d_ws, size_t ws_size,
                              hipStream_t stream) {
    const float* x   = (const float*)d_in[0];   // [T,D]
    const float* Wr1 = (const float*)d_in[1];   // [D,D]
    const float* br1 = (const float*)d_in[2];   // [D]
    const float* Wr2 = (const float*)d_in[3];   // [D,E]
    const float* br2 = (const float*)d_in[4];   // [E]
    const float* W1  = (const float*)d_in[5];   // [E,D,H]
    const float* b1  = (const float*)d_in[6];   // [E,H]
    const float* W2  = (const float*)d_in[7];   // [E,H,D]
    const float* b2  = (const float*)d_in[8];   // [E,D]
    float* out = (float*)d_out;

    const int D = in_sizes[2];            // 1024
    const int E = in_sizes[4];            // 8
    const int H = in_sizes[6] / E;        // 4096
    const int T = in_sizes[0] / D;        // 4096

    // workspace layout (bf16 unless noted)
    unsigned short* xb   = (unsigned short*)d_ws;          // [T,D]
    unsigned short* xh   = xb   + (size_t)T * D;           // [T,D]
    unsigned short* hbuf = xh   + (size_t)T * D;           // [T,H]
    unsigned short* Wr1t = hbuf + (size_t)T * H;           // [D,D]
    unsigned short* W1t  = Wr1t + (size_t)D * D;           // [E][H,D] (transposed)
    unsigned short* W2t  = W1t  + (size_t)E * D * H;       // [E][D,H] (transposed)
    float*          rwp  = (float*)(W2t + (size_t)E * H * D);  // [T,E] fp32

    dim3 blk(256), tblk(32, 8);

    // pre-pass: convert/transpose to bf16 (amortized, memory-bound)
    cvt_bf16<<<dim3((T * D / 4 + 255) / 256), blk, 0, stream>>>(x, xb, T * D / 4);
    transpose_cvt<<<dim3(D / 32, D / 32, 1), tblk, 0, stream>>>(Wr1, Wr1t, D, D);
    transpose_cvt<<<dim3(H / 32, D / 32, E), tblk, 0, stream>>>(W1, W1t, D, H);
    transpose_cvt<<<dim3(D / 32, H / 32, E), tblk, 0, stream>>>(W2, W2t, H, D);

    // 1) router hidden: xh = relu(x @ Wr1 + br1)   (bf16 out)
    gemm_bf16_wmma<<<dim3(D / BN, T / BM), blk, 0, stream>>>(
        xb, Wr1t, br1, xh, nullptr, 0, 0, D, D, /*epi=*/0);

    // 2) rw = softmax(xh @ Wr2 + br2)
    router_softmax<<<dim3(T / 8), blk, 0, stream>>>(xh, Wr2, br2, rwp, T, D);

    // 3) experts (stream-serialized -> deterministic accumulation into out)
    for (int e = 0; e < E; ++e) {
        gemm_bf16_wmma<<<dim3(H / BN, T / BM), blk, 0, stream>>>(
            xb, W1t + (size_t)e * D * H, b1 + (size_t)e * H, hbuf,
            nullptr, 0, 0, H, D, /*epi=*/1);
        gemm_bf16_wmma<<<dim3(D / BN, T / BM), blk, 0, stream>>>(
            hbuf, W2t + (size_t)e * H * D, b2 + (size_t)e * D, out,
            rwp, e, /*accum=*/(e > 0), D, H, /*epi=*/2);
    }
}
